// ChiralMessage_53395033423975
// MI455X (gfx1250) — compile-verified, matrix operand order checked by reading, not run on hardware
//
#include <hip/hip_runtime.h>
#include <hip/hip_bf16.h>
#include <math.h>

typedef __bf16 bf16_t;
typedef __attribute__((ext_vector_type(16))) __bf16 v16bf;
typedef __attribute__((ext_vector_type(8)))  __bf16 v8bf;
typedef __attribute__((ext_vector_type(8)))  float  v8f;

#define F 64
#define PI_F 3.14159265358979323846f
#define CUTOFF_F 5.0f

__device__ __forceinline__ float fsign(float x){ return (x>0.f)?1.f:((x<0.f)?-1.f:0.f); }
__device__ __forceinline__ float fast_rcp(float x){ return __builtin_amdgcn_rcpf(x); }
__device__ __forceinline__ float silu_f(float x){ return x * fast_rcp(1.0f + __expf(-x)); }

__device__ __forceinline__ void atomicMaxF(float* a, float v){
  if (v >= 0.f) atomicMax((int*)a, __float_as_int(v));
  else          atomicMin((unsigned int*)a, __float_as_uint(v));
}
__device__ __forceinline__ void atomicMinF(float* a, float v){
  if (v >= 0.f) atomicMin((int*)a, __float_as_int(v));
  else          atomicMax((unsigned int*)a, __float_as_uint(v));
}

// Stage W[K][NOUT] (global f32) -> Wt[NOUT][K+8] (LDS bf16, transposed, padded).
template<int K, int NOUT>
__device__ __forceinline__ void stage_wt(const float* __restrict__ Wg, bf16_t* __restrict__ Wt,
                                         int tid, int nthr)
{
  constexpr int LD = K + 8;
  for (int i = tid; i < K*NOUT; i += nthr){
    int k = i / NOUT, c = i - k*NOUT;
    Wt[c*LD + k] = (bf16_t)Wg[i];
  }
}

// One wave computes Y[16][NOUT] = (opt silu)(X[16][K] @ W[K][NOUT] + bias).
// X: bf16 [16][K+8] LDS (row-major, padded). Wt: bf16 [NOUT][K+8] LDS (transposed, padded).
// out: f32 [16][NOUT] LDS. Fragments fetched as 16-byte LDS vector loads (ds_load_b128);
// layout per CDNA5 ISA 16-bit A 16x32 (lane = row | khalf), C/D 16x16 f32.
template<int K, int NOUT, bool SILU>
__device__ __forceinline__ void wave_gemm(const bf16_t* __restrict__ X,
                                          const bf16_t* __restrict__ Wt,
                                          const float*  __restrict__ bias,
                                          float* __restrict__ out, int lane)
{
  constexpr int LD = K + 8;
  const int row = lane & 15;
  const int kh  = lane >> 4;
  union Frag { v16bf v; v8bf h[2]; };
  const bf16_t* xr = X + row*LD + kh*8;
  Frag a[K/32];
#pragma unroll
  for (int kk = 0; kk < K/32; ++kk){
    a[kk].h[0] = *(const v8bf*)(xr + kk*32);
    a[kk].h[1] = *(const v8bf*)(xr + kk*32 + 16);
  }
#pragma unroll
  for (int n0 = 0; n0 < NOUT; n0 += 16) {
    const bf16_t* wr = Wt + (n0 + row)*LD + kh*8;
    v8f acc = {};
#pragma unroll
    for (int kk = 0; kk < K/32; ++kk) {
      Frag b;
      b.h[0] = *(const v8bf*)(wr + kk*32);
      b.h[1] = *(const v8bf*)(wr + kk*32 + 16);
      acc = __builtin_amdgcn_wmma_f32_16x16x32_bf16(false, a[kk].v, false, b.v,
                                                    (short)0, acc, false, false);
    }
    float bv = bias[n0 + row];
#pragma unroll
    for (int v = 0; v < 8; ++v) {
      float val = acc[v] + bv;
      if (SILU) val = silu_f(val);
      out[(kh*8 + v)*NOUT + n0 + row] = val;
    }
  }
}

// ---------------- init: zero accumulators, seed outputs ----------------
__global__ void k_init(const float* __restrict__ nch, const float* __restrict__ nvec,
                       float* mss, float* esum, float* emax, float* emin, float* cnt,
                       float* out1, float* out2, int N)
{
  long i = (long)blockIdx.x*blockDim.x + threadIdx.x;
  long n64 = (long)N*64, n192 = (long)N*192;
  if (i < n64){
    mss[i]=0.f; esum[i]=0.f;
    emax[i]=-__builtin_huge_valf(); emin[i]=__builtin_huge_valf();
    out1[i]=nch[i];
  }
  if (i < n192) out2[i]=nvec[i];
  if (i < N) cnt[i]=0.f;
}

// ---------------- edge kernel: gate MLP + filter + softmax + scatter ----------------
__global__ __launch_bounds__(128) void k_edge(
    const float* __restrict__ ns, const float* __restrict__ nvec,
    const int* __restrict__ ei, const float* __restrict__ ediff, const float* __restrict__ edist,
    const float* __restrict__ sg1_w, const float* __restrict__ sg1_b,
    const float* __restrict__ sg2_w, const float* __restrict__ sg2_b,
    const float* __restrict__ sfilt_w, const float* __restrict__ sfilt_b,
    float* mss, float* esum, float* emax, float* emin, float* cnt, float* out2, int E)
{
  __shared__ bf16_t w1t[64*136];     // sg1^T: [64][128+8]
  __shared__ bf16_t w2t[256*72];     // sg2^T: [256][64+8]
  __shared__ bf16_t xs[4][16*136];
  __shared__ bf16_t hb[4][16*72];
  __shared__ float  gt[4][16*256];
  __shared__ float  sc[4][16*6];

  const int tid = threadIdx.x, lane = tid & 31, w = tid >> 5;
  stage_wt<128,64>(sg1_w, w1t, tid, 128);
  stage_wt<64,256>(sg2_w, w2t, tid, 128);
  __syncthreads();

  const int ebase = blockIdx.x*64 + w*16;
  __builtin_prefetch(&ediff[(long)ebase*3], 0, 1);

  // gather X = [ns[src] | ns[dst]] as bf16
  for (int idx = lane; idx < 16*128; idx += 32){
    int r = idx >> 7, c = idx & 127;
    int e = ebase + r;
    float v = 0.f;
    if (e < E){
      int node = (c < 64) ? ei[e*2] : ei[e*2+1];
      v = ns[(long)node*64 + (c & 63)];
    }
    xs[w][r*136 + c] = (bf16_t)v;
  }
  __syncthreads();
  wave_gemm<128,64,true>(xs[w], w1t, sg1_b, gt[w], lane);        // hidden (silu)
  __syncthreads();
  for (int idx = lane; idx < 16*64; idx += 32){
    int r = idx >> 6, f = idx & 63;
    hb[w][r*72 + f] = (bf16_t)gt[w][idx];
  }
  __syncthreads();
  wave_gemm<64,256,false>(hb[w], w2t, sg2_b, gt[w], lane);       // gate 16x256

  // sinc expansion + cosine cutoff per edge
  if (lane < 16){
    int e = ebase + lane;
    float d = (e < E) ? edist[e] : 1.f;
    float rd = fast_rcp(d);
#pragma unroll
    for (int n = 1; n <= 5; ++n) sc[w][lane*6 + n-1] = __sinf(d * n * (PI_F/CUTOFF_F)) * rd;
    sc[w][lane*6 + 5] = (d < CUTOFF_F) ? 0.5f*(__cosf(PI_F*d/CUTOFF_F)+1.f) : 0.f;
  }
  __syncthreads();
  // gate *= (sinc @ sfilt + b) * cutoff
  for (int idx = lane; idx < 16*256; idx += 32){
    int e = idx >> 8, f = idx & 255;
    float filt = sfilt_b[f];
#pragma unroll
    for (int r5 = 0; r5 < 5; ++r5) filt += sc[w][e*6+r5] * sfilt_w[r5*256+f];
    gt[w][idx] *= filt * sc[w][e*6+5];
  }
  __syncthreads();
  // softmax over gate_vv (cols 0..63) and gate_ev (cols 64..127)
  {
    int e = lane & 15, seg = lane >> 4;
    float* p = &gt[w][e*256 + seg*64];
    float m = -__builtin_huge_valf();
    for (int f = 0; f < 64; ++f) m = fmaxf(m, p[f]);
    float s = 0.f;
    for (int f = 0; f < 64; ++f){ float ev = __expf(p[f]-m); p[f]=ev; s+=ev; }
    float inv = fast_rcp(s);
    for (int f = 0; f < 64; ++f) p[f] *= inv;
  }
  __syncthreads();
  // scatter (segment reductions via L2 atomics)
  for (int idx = lane; idx < 16*64; idx += 32){
    int e = idx >> 6, f = idx & 63;
    int ge = ebase + e;
    if (ge >= E) continue;
    int src = ei[ge*2], dst = ei[ge*2+1];
    float gvv  = gt[w][e*256 + f];
    float gev  = gt[w][e*256 + 64 + f];
    float m_ss = gt[w][e*256 + 128 + f];
    float envc = gt[w][e*256 + 192 + f];
    atomicAdd(&mss [(long)src*64+f], m_ss);
    atomicAdd(&esum[(long)src*64+f], envc);
    atomicMaxF(&emax[(long)src*64+f], envc);
    atomicMinF(&emin[(long)src*64+f], envc);
    if (f == 0) atomicAdd(&cnt[src], 1.f);
#pragma unroll
    for (int d = 0; d < 3; ++d){
      float md = gvv * nvec[(long)dst*192 + d*64 + f] + gev * ediff[(long)ge*3 + d];
      atomicAdd(&out2[(long)src*192 + d*64 + f], md);
    }
  }
}

// ---------------- env MLP: env = MLP2([sum|max|min]) ----------------
__global__ __launch_bounds__(128) void k_env(
    const float* __restrict__ esum, const float* __restrict__ emax,
    const float* __restrict__ emin, const float* __restrict__ cnt,
    const float* __restrict__ env1_w, const float* __restrict__ env1_b,
    const float* __restrict__ env2_w, const float* __restrict__ env2_b,
    float* envv, int N)
{
  __shared__ bf16_t w1t[64*200];     // env1^T: [64][192+8]
  __shared__ bf16_t w2t[64*72];      // env2^T
  __shared__ bf16_t xb[4][16*200];
  __shared__ float  hf[4][16*64];
  __shared__ bf16_t hb[4][16*72];
  const int tid=threadIdx.x, lane=tid&31, w=tid>>5;
  stage_wt<192,64>(env1_w, w1t, tid, 128);
  stage_wt<64,64>(env2_w, w2t, tid, 128);
  __syncthreads();
  int nbase = blockIdx.x*64 + w*16;
  for (int r=0;r<16;++r){
    int n = nbase + r;
    bool valid = n < N;
    float c = valid ? cnt[n] : 0.f;
    for (int col = lane; col < 192; col += 32){
      float v = 0.f;
      if (valid){
        if (col < 64)       v = esum[(long)n*64+col];
        else if (col < 128) v = (c>0.f)? emax[(long)n*64+col-64]  : 0.f;
        else                v = (c>0.f)? emin[(long)n*64+col-128] : 0.f;
      }
      xb[w][r*200+col] = (bf16_t)v;
    }
  }
  __syncthreads();
  wave_gemm<192,64,true>(xb[w], w1t, env1_b, hf[w], lane);
  __syncthreads();
  for (int idx=lane;idx<16*64;idx+=32){
    int r=idx>>6, f=idx&63;
    hb[w][r*72+f]=(bf16_t)hf[w][idx];
  }
  __syncthreads();
  wave_gemm<64,64,false>(hb[w], w2t, env2_b, hf[w], lane);
  __syncthreads();
  for (int idx=lane; idx<16*64; idx+=32){
    int r=idx>>6, f=idx&63; int n=nbase+r;
    if (n<N) envv[(long)n*64+f]=hf[w][idx];
  }
}

// ---------------- triplet kernel: chiral messages ----------------
__global__ __launch_bounds__(64) void k_trip(
    const float* __restrict__ ns, const float* __restrict__ nvec,
    const float* __restrict__ pos, const int* __restrict__ tri,
    const float* __restrict__ ece_w,  const float* __restrict__ ece_b,
    const float* __restrict__ ecv1_w, const float* __restrict__ ecv1_b,
    const float* __restrict__ ecv2_w, const float* __restrict__ ecv2_b,
    const float* __restrict__ cvf1_w, const float* __restrict__ cvf1_b,
    const float* __restrict__ cvf2_w, const float* __restrict__ cvf2_b,
    const float* __restrict__ cee1_w, const float* __restrict__ cee1_b,
    const float* __restrict__ cee2_w, const float* __restrict__ cee2_b,
    const float* __restrict__ cve1_w, const float* __restrict__ cve1_b,
    const float* __restrict__ cve2_w, const float* __restrict__ cve2_b,
    const float* __restrict__ envv, float* out1, int T)
{
  __shared__ bf16_t W_ece[64*72], W_ecv1[64*72], W_ecv2[64*72], W_cvf1[64*72], W_cvf2[64*72];
  __shared__ bf16_t W_cee1[64*136], W_cve1[64*136];
  __shared__ bf16_t sb[2][16*72];
  __shared__ bf16_t s123b[2][16*72];
  __shared__ bf16_t catb[2][16*136];
  __shared__ bf16_t hbw[2][16*72];
  __shared__ float  hf[2][16*64];
  __shared__ float  cef[2][16*64];
  __shared__ float  cvv[2][16*64];
  __shared__ float  cvfv[2][16*64];
  __shared__ int    tix[2][64];
  __shared__ float  cee_s[2][16], cve_s[2][16], mterm_s[2][16], sgn_s[2][16];

  const int tid=threadIdx.x, lane=tid&31, w=tid>>5;
  stage_wt<64,64>(ece_w,  W_ece,  tid, 64);
  stage_wt<64,64>(ecv1_w, W_ecv1, tid, 64);
  stage_wt<64,64>(ecv2_w, W_ecv2, tid, 64);
  stage_wt<64,64>(cvf1_w, W_cvf1, tid, 64);
  stage_wt<64,64>(cvf2_w, W_cvf2, tid, 64);
  stage_wt<128,64>(cee1_w, W_cee1, tid, 64);
  stage_wt<128,64>(cve1_w, W_cve1, tid, 64);
  __syncthreads();

  const int tbase = blockIdx.x*32 + w*16;
  for (int i=lane;i<64;i+=32){
    int row = tbase + (i>>2);
    tix[w][i] = (row < T) ? tri[(long)tbase*4 + i] : 0;
  }
  __syncthreads();
  // ssum, s123, env[b]
  for (int idx=lane; idx<16*64; idx+=32){
    int r=idx>>6, f=idx&63;
    int ib=tix[w][r*4], i1=tix[w][r*4+1], i2=tix[w][r*4+2], i3=tix[w][r*4+3];
    float s3 = ns[(long)i1*64+f] + ns[(long)i2*64+f] + ns[(long)i3*64+f];
    s123b[w][r*72+f] = (bf16_t)s3;
    sb[w][r*72+f]    = (bf16_t)(s3 + ns[(long)ib*64+f]);
    catb[w][r*136+f] = (bf16_t)envv[(long)ib*64+f];
  }
  __syncthreads();
  wave_gemm<64,64,true>(sb[w], W_ece, ece_b, cef[w], lane);        // chiral_edge
  __syncthreads();
  for (int idx=lane;idx<16*64;idx+=32){ int r=idx>>6,f=idx&63; catb[w][r*136+64+f]=(bf16_t)cef[w][idx]; }
  __syncthreads();
  wave_gemm<128,64,true>(catb[w], W_cee1, cee1_b, hf[w], lane);    // cee hidden
  __syncthreads();
  if (lane<16){
    float s = cee2_b[0];
    for (int f=0;f<64;++f) s += hf[w][lane*64+f]*cee2_w[f];
    cee_s[w][lane]=s;
  }
  __syncthreads();
  wave_gemm<64,64,true>(sb[w], W_ecv1, ecv1_b, hf[w], lane);       // chiral_vector MLP
  __syncthreads();
  for (int idx=lane;idx<16*64;idx+=32){ int r=idx>>6,f=idx&63; hbw[w][r*72+f]=(bf16_t)hf[w][idx]; }
  __syncthreads();
  wave_gemm<64,64,false>(hbw[w], W_ecv2, ecv2_b, cvv[w], lane);
  __syncthreads();
  for (int idx=lane;idx<16*64;idx+=32){ int r=idx>>6,f=idx&63; catb[w][r*136+64+f]=(bf16_t)cvv[w][idx]; }
  __syncthreads();
  wave_gemm<128,64,true>(catb[w], W_cve1, cve1_b, hf[w], lane);    // cve hidden
  __syncthreads();
  if (lane<16){
    float s = cve2_b[0];
    for (int f=0;f<64;++f) s += hf[w][lane*64+f]*cve2_w[f];
    cve_s[w][lane]=s;
  }
  __syncthreads();
  wave_gemm<64,64,true>(s123b[w], W_cvf1, cvf1_b, hf[w], lane);    // cvf MLP
  __syncthreads();
  for (int idx=lane;idx<16*64;idx+=32){ int r=idx>>6,f=idx&63; hbw[w][r*72+f]=(bf16_t)hf[w][idx]; }
  __syncthreads();
  wave_gemm<64,64,false>(hbw[w], W_cvf2, cvf2_b, cvfv[w], lane);
  __syncthreads();
  // geometry: 2 lanes per triplet, split a-range
  {
    int r = lane & 15, p = lane >> 4;
    int ib=tix[w][r*4], i1=tix[w][r*4+1], i2=tix[w][r*4+2], i3=tix[w][r*4+3];
    float r1[3], r2[3], r3[3];
#pragma unroll
    for (int d=0;d<3;++d){
      float pbv = pos[(long)ib*3+d];
      r1[d]=pbv-pos[(long)i1*3+d];
      r2[d]=pbv-pos[(long)i2*3+d];
      r3[d]=pbv-pos[(long)i3*3+d];
    }
    float c0 = r2[1]*r3[2]-r2[2]*r3[1];
    float c1 = r2[2]*r3[0]-r2[0]*r3[2];
    float c2 = r2[0]*r3[1]-r2[1]*r3[0];
    float tp = r1[0]*c0+r1[1]*c1+r1[2]*c2;
    float v1a[4][3], v2a[4][3], v3a[4][3];
#pragma unroll
    for (int a=0;a<4;++a)
#pragma unroll
      for (int d=0;d<3;++d){
        v1a[a][d]=r1[d]+nvec[(long)i1*192+d*64+a];
        v2a[a][d]=r2[d]+nvec[(long)i2*192+d*64+a];
        v3a[a][d]=r3[d]+nvec[(long)i3*192+d*64+a];
      }
    float part = 0.f;
    for (int bb=0;bb<4;++bb)
      for (int cc=0;cc<4;++cc){
        float x0=v2a[bb][1]*v3a[cc][2]-v2a[bb][2]*v3a[cc][1];
        float x1=v2a[bb][2]*v3a[cc][0]-v2a[bb][0]*v3a[cc][2];
        float x2=v2a[bb][0]*v3a[cc][1]-v2a[bb][1]*v3a[cc][0];
        for (int a=p*2; a<p*2+2; ++a){
          float dp = v1a[a][0]*x0+v1a[a][1]*x1+v1a[a][2]*x2;
          part += fsign(dp)*cvfv[w][r*64 + a*16 + bb*4 + cc];
        }
      }
    part += __shfl_xor(part, 16, 32);
    if (p==0){ mterm_s[w][r]=part*(1.f/64.f); sgn_s[w][r]=fsign(tp); }
  }
  __syncthreads();
  // scatter chiral message to out1
  for (int idx=lane; idx<16*64; idx+=32){
    int r=idx>>6, f=idx&63;
    if (tbase + r >= T) continue;
    int ib=tix[w][r*4];
    float val = cef[w][idx]*cee_s[w][r]*sgn_s[w][r]
              + cvv[w][idx]*mterm_s[w][r]*cve_s[w][r];
    atomicAdd(&out1[(long)ib*64+f], val);
  }
}

// ---------------- node update: updV norm, svg, su ----------------
__global__ __launch_bounds__(128) void k_node(
    const float* __restrict__ ns, const float* __restrict__ nvec, const float* __restrict__ mss,
    const float* __restrict__ updV_w, const float* __restrict__ updV_b,
    const float* __restrict__ svg1_w, const float* __restrict__ svg1_b,
    const float* __restrict__ svg2_w, const float* __restrict__ svg2_b,
    const float* __restrict__ su1_w,  const float* __restrict__ su1_b,
    const float* __restrict__ su2_w,  const float* __restrict__ su2_b,
    float* out0, int N)
{
  __shared__ bf16_t Wv[64*72], Ws1[64*72], Ws2[64*72], Wu1[64*72], Wu2[64*72];
  __shared__ bf16_t xb[4][16*72];
  __shared__ bf16_t tb[4][16*72];
  __shared__ bf16_t hbw[4][16*72];
  __shared__ float  hf[4][16*64];
  __shared__ float  vn2[4][16*64];
  __shared__ float  mv[4][16*64];
  const int tid=threadIdx.x, lane=tid&31, w=tid>>5;
  stage_wt<64,64>(updV_w, Wv,  tid, 128);
  stage_wt<64,64>(svg1_w, Ws1, tid, 128);
  stage_wt<64,64>(svg2_w, Ws2, tid, 128);
  stage_wt<64,64>(su1_w,  Wu1, tid, 128);
  stage_wt<64,64>(su2_w,  Wu2, tid, 128);
  __syncthreads();
  int nbase = blockIdx.x*64 + w*16;
  for (int idx=lane; idx<16*64; idx+=32){
    int r=idx>>6, f=idx&63; int n=nbase+r;
    xb[w][r*72+f] = (bf16_t)((n<N)? ns[(long)n*64+f] : 0.f);
    vn2[w][idx] = 0.f;
  }
  __syncthreads();
  for (int d=0; d<3; ++d){
    for (int idx=lane; idx<16*64; idx+=32){
      int r=idx>>6, f=idx&63; int n=nbase+r;
      tb[w][r*72+f] = (bf16_t)((n<N)? nvec[(long)n*192 + d*64 + f] : 0.f);
    }
    __syncthreads();
    wave_gemm<64,64,false>(tb[w], Wv, updV_b, hf[w], lane);
    __syncthreads();
    for (int i=lane;i<16*64;i+=32){ float v=hf[w][i]; vn2[w][i]+=v*v; }
    __syncthreads();
  }
  wave_gemm<64,64,true>(xb[w], Ws1, svg1_b, hf[w], lane);
  __syncthreads();
  for (int idx=lane;idx<16*64;idx+=32){ int r=idx>>6,f=idx&63; hbw[w][r*72+f]=(bf16_t)hf[w][idx]; }
  __syncthreads();
  wave_gemm<64,64,false>(hbw[w], Ws2, svg2_b, mv[w], lane);
  __syncthreads();
  // z = ns + message_vs + message_ss
  for (int idx=lane; idx<16*64; idx+=32){
    int r=idx>>6, f=idx&63; int n=nbase+r;
    float z = 0.f;
    if (n < N) z = ns[(long)n*64+f] + mv[w][idx]*sqrtf(vn2[w][idx]) + mss[(long)n*64+f];
    xb[w][r*72+f] = (bf16_t)z;
  }
  __syncthreads();
  wave_gemm<64,64,true>(xb[w], Wu1, su1_b, hf[w], lane);
  __syncthreads();
  for (int idx=lane;idx<16*64;idx+=32){ int r=idx>>6,f=idx&63; hbw[w][r*72+f]=(bf16_t)hf[w][idx]; }
  __syncthreads();
  wave_gemm<64,64,false>(hbw[w], Wu2, su2_b, mv[w], lane);
  __syncthreads();
  for (int idx=lane; idx<16*64; idx+=32){
    int r=idx>>6, f=idx&63; int n=nbase+r;
    if (n<N) out0[(long)n*64+f]=mv[w][idx];
  }
}

extern "C" void kernel_launch(void* const* d_in, const int* in_sizes, int n_in,
                              void* d_out, int out_size, void* d_ws, size_t ws_size,
                              hipStream_t stream)
{
  (void)n_in; (void)out_size; (void)ws_size;
  const float* ns    = (const float*)d_in[0];
  const float* nch   = (const float*)d_in[1];
  const float* nvec  = (const float*)d_in[2];
  const int*   ei    = (const int*)d_in[3];
  const float* ediff = (const float*)d_in[4];
  const float* edist = (const float*)d_in[5];
  const int*   tri   = (const int*)d_in[6];
  const float* pos   = (const float*)d_in[7];
  // params dict flattened as pytree leaves (keys sorted), each param = (W, b)
  const float* cee1_w=(const float*)d_in[8];  const float* cee1_b=(const float*)d_in[9];
  const float* cee2_w=(const float*)d_in[10]; const float* cee2_b=(const float*)d_in[11];
  const float* cve1_w=(const float*)d_in[12]; const float* cve1_b=(const float*)d_in[13];
  const float* cve2_w=(const float*)d_in[14]; const float* cve2_b=(const float*)d_in[15];
  const float* cvf1_w=(const float*)d_in[16]; const float* cvf1_b=(const float*)d_in[17];
  const float* cvf2_w=(const float*)d_in[18]; const float* cvf2_b=(const float*)d_in[19];
  const float* ece_w =(const float*)d_in[20]; const float* ece_b =(const float*)d_in[21];
  const float* ecv1_w=(const float*)d_in[22]; const float* ecv1_b=(const float*)d_in[23];
  const float* ecv2_w=(const float*)d_in[24]; const float* ecv2_b=(const float*)d_in[25];
  const float* env1_w=(const float*)d_in[26]; const float* env1_b=(const float*)d_in[27];
  const float* env2_w=(const float*)d_in[28]; const float* env2_b=(const float*)d_in[29];
  const float* sfilt_w=(const float*)d_in[30];const float* sfilt_b=(const float*)d_in[31];
  const float* sg1_w =(const float*)d_in[32]; const float* sg1_b =(const float*)d_in[33];
  const float* sg2_w =(const float*)d_in[34]; const float* sg2_b =(const float*)d_in[35];
  const float* su1_w =(const float*)d_in[36]; const float* su1_b =(const float*)d_in[37];
  const float* su2_w =(const float*)d_in[38]; const float* su2_b =(const float*)d_in[39];
  const float* svg1_w=(const float*)d_in[40]; const float* svg1_b=(const float*)d_in[41];
  const float* svg2_w=(const float*)d_in[42]; const float* svg2_b=(const float*)d_in[43];
  const float* updV_w=(const float*)d_in[44]; const float* updV_b=(const float*)d_in[45];

  const int N = in_sizes[0] / F;
  const int E = in_sizes[5];
  const int T = in_sizes[6] / 4;

  long n64 = (long)N * F;
  float* mss  = (float*)d_ws;
  float* esum = mss  + n64;
  float* emax = esum + n64;
  float* emin = emax + n64;
  float* envv = emin + n64;
  float* cnt  = envv + n64;

  float* out0 = (float*)d_out;
  float* out1 = out0 + n64;
  float* out2 = out1 + n64;

  {
    long tot = (long)N * 192;
    int blocks = (int)((tot + 255) / 256);
    k_init<<<blocks, 256, 0, stream>>>(nch, nvec, mss, esum, emax, emin, cnt, out1, out2, N);
  }
  k_edge<<<(E + 63) / 64, 128, 0, stream>>>(ns, nvec, ei, ediff, edist,
      sg1_w, sg1_b, sg2_w, sg2_b, sfilt_w, sfilt_b,
      mss, esum, emax, emin, cnt, out2, E);
  k_env<<<(N + 63) / 64, 128, 0, stream>>>(esum, emax, emin, cnt,
      env1_w, env1_b, env2_w, env2_b, envv, N);
  k_trip<<<(T + 31) / 32, 64, 0, stream>>>(ns, nvec, pos, tri,
      ece_w, ece_b, ecv1_w, ecv1_b, ecv2_w, ecv2_b,
      cvf1_w, cvf1_b, cvf2_w, cvf2_b,
      cee1_w, cee1_b, cee2_w, cee2_b,
      cve1_w, cve1_b, cve2_w, cve2_b,
      envv, out1, T);
  k_node<<<(N + 63) / 64, 128, 0, stream>>>(ns, nvec, mss,
      updV_w, updV_b, svg1_w, svg1_b, svg2_w, svg2_b,
      su1_w, su1_b, su2_w, su2_b, out0, N);
}
